// GAT_LSTM_22832046146096
// MI455X (gfx1250) — matrix operand order repaired and checked by reading, not compiled, not run on hardware
//
#include <hip/hip_runtime.h>

// ---------------- types & helpers ----------------
typedef __attribute__((ext_vector_type(16))) __bf16 v16bf;
typedef __attribute__((ext_vector_type(8)))  __bf16 v8bf;
typedef __attribute__((ext_vector_type(8)))  float  v8f;

#define B_    4
#define P_    12
#define N_    1000
#define NP_   1008      // N padded to multiple of 16
#define H_    128
#define FUT_  6
#define BP_   (B_ * P_) // 48
#define TILES_ 63       // NP_/16
#define BN_   (B_ * N_) // 4000 LSTM rows
#define BNT_  250       // BN_/16

__device__ __forceinline__ __bf16 f2bf(float f) {
  unsigned u = __builtin_bit_cast(unsigned, f);
  unsigned r = (u + 0x7fffu + ((u >> 16) & 1u)) >> 16;
  unsigned short s = (unsigned short)r;
  return __builtin_bit_cast(__bf16, s);
}
__device__ __forceinline__ float sigmf(float x) { return 1.f / (1.f + __expf(-x)); }

// A-matrix fragment, 16x32 bf16, row-major source with leading dim ld.
// Lane L: row = L&15; elems 0..15 -> K = (L>>4)*16+e
__device__ __forceinline__ v16bf load_a(const __bf16* base, int ld, int kb) {
  int lane = threadIdx.x & 31;
  int row  = lane & 15;
  int half = lane >> 4;
  const __bf16* p = base + (size_t)row * ld + kb * 32 + half * 16;
  union { v16bf v; uint4 q[2]; } u;
  u.q[0] = *(const uint4*)p;
  u.q[1] = *(const uint4*)(p + 8);
  return u.v;
}

// B-matrix fragment from swizzled storage: frag = (kb * nColTiles + ct);
// layout [frag][lane][16 contiguous elems] -> two b128 loads per lane.
__device__ __forceinline__ v16bf load_bsw(const __bf16* w, int frag) {
  int lane = threadIdx.x & 31;
  const __bf16* p = w + ((size_t)frag * 32 + lane) * 16;
  union { v16bf v; uint4 q[2]; } u;
  u.q[0] = *(const uint4*)p;
  u.q[1] = *(const uint4*)(p + 8);
  return u.v;
}

#define WMMA(acc, a, b) \
  (acc) = __builtin_amdgcn_wmma_f32_16x16x32_bf16(false, (a), false, (b), (short)0, (acc), false, false)

__device__ __forceinline__ v8f v8zero() {
  v8f z = {0.f, 0.f, 0.f, 0.f, 0.f, 0.f, 0.f, 0.f};
  return z;
}

// one K-step against 8 column tiles: batch-load all B frags, then 8 WMMAs.
// Distinct B registers let the compiler issue one load clause and overlap
// WMMA issue with outstanding loads (progressive s_wait_loadcnt).
__device__ __forceinline__ void kstep8(v8f acc[8], v16bf a, const __bf16* w, int fbase) {
  v16bf bf[8];
#pragma unroll
  for (int ct = 0; ct < 8; ++ct) bf[ct] = load_bsw(w, fbase + ct);
#pragma unroll
  for (int ct = 0; ct < 8; ++ct) WMMA(acc[ct], a, bf[ct]);
}

// ---------------- weight fp32 -> bf16, swizzled into B-fragment order ----------------
__global__ void k_convw_sw(const float* __restrict__ src, __bf16* __restrict__ dst,
                           int K, int Kpad, int N) {
  int idx = blockIdx.x * blockDim.x + threadIdx.x;
  if (idx >= Kpad * N) return;
  int e    = idx & 15;
  int lane = (idx >> 4) & 31;
  int fr   = idx >> 9;
  int nct  = N >> 4;
  int ct   = fr % nct;
  int kb   = fr / nct;
  int k = kb * 32 + ((lane >> 4) << 4) + e;
  int n = ct * 16 + (lane & 15);
  dst[idx] = (k < K) ? f2bf(src[(size_t)k * N + n]) : f2bf(0.f);
}

// ---------------- build embedded input (B,P,NP_,160) bf16, zero-padded ----------------
__global__ void k_embed(const float* __restrict__ x, const float* __restrict__ E0,
                        const float* __restrict__ E1, __bf16* __restrict__ Xemb) {
  int t = blockIdx.x;            // bp*NP_ + row
  int f = threadIdx.x;           // 0..159
  int row = t % NP_;
  int bp  = t / NP_;
  float v = 0.f;
  if (row < N_) {
    const float* xr = x + ((size_t)bp * N_ + row) * 10;
    if (f < 8)        v = xr[f];
    else if (f < 72)  { int i0 = (int)xr[8]; v = E0[i0 * 64 + (f - 8)]; }
    else if (f < 136) { int i1 = (int)xr[9]; v = E1[i1 * 64 + (f - 72)]; }
  }
  Xemb[(size_t)t * 160 + f] = f2bf(v);
}

// ---------------- fused MLP: W1/relu, W2/relu, W3 -> h(swizzled) ; We -> s1,s2 ----------------
__global__ __launch_bounds__(32) void k_mlp(
    const __bf16* __restrict__ Xemb, const __bf16* __restrict__ W1s,
    const __bf16* __restrict__ W2s,  const __bf16* __restrict__ W3s,
    const __bf16* __restrict__ Wes,
    const float* __restrict__ b1, const float* __restrict__ b2,
    const float* __restrict__ a1, const float* __restrict__ a2,
    __bf16* __restrict__ hsw, float* __restrict__ s1, float* __restrict__ s2) {
  __shared__ __align__(16) __bf16 t0[16 * H_];
  __shared__ __align__(16) __bf16 t1[16 * H_];
  __shared__ __align__(16) float  xs[16 * H_];
  int bp = blockIdx.x / TILES_;
  int tile = blockIdx.x % TILES_;
  int rowbase = tile * 16;
  int lane = threadIdx.x, colL = lane & 15, half = lane >> 4;

  v8f acc[8];
  // ---- layer 1: K=160 ----
#pragma unroll
  for (int ct = 0; ct < 8; ++ct) acc[ct] = v8zero();
  const __bf16* Abase = Xemb + ((size_t)bp * NP_ + rowbase) * 160;
#pragma unroll
  for (int kb = 0; kb < 5; ++kb) {
    v16bf a = load_a(Abase, 160, kb);
    kstep8(acc, a, W1s, kb * 8);
  }
#pragma unroll
  for (int ct = 0; ct < 8; ++ct) {
    int col = ct * 16 + colL; float bias = b1[col];
#pragma unroll
    for (int v = 0; v < 8; ++v) {
      float val = acc[ct][v] + bias;
      t0[(half * 8 + v) * H_ + col] = f2bf(val > 0.f ? val : 0.f);
    }
  }
  // ---- layer 2: K=128 ----
#pragma unroll
  for (int ct = 0; ct < 8; ++ct) acc[ct] = v8zero();
#pragma unroll
  for (int kb = 0; kb < 4; ++kb) {
    v16bf a = load_a(t0, H_, kb);
    kstep8(acc, a, W2s, kb * 8);
  }
#pragma unroll
  for (int ct = 0; ct < 8; ++ct) {
    int col = ct * 16 + colL; float bias = b2[col];
#pragma unroll
    for (int v = 0; v < 8; ++v) {
      float val = acc[ct][v] + bias;
      t1[(half * 8 + v) * H_ + col] = f2bf(val > 0.f ? val : 0.f);
    }
  }
  // ---- layer 3: W3 (no activation) -> h ----
#pragma unroll
  for (int ct = 0; ct < 8; ++ct) acc[ct] = v8zero();
#pragma unroll
  for (int kb = 0; kb < 4; ++kb) {
    v16bf a = load_a(t1, H_, kb);
    kstep8(acc, a, W3s, kb * 8);
  }
  // store h: row-major bf16 into t0 (feeds We), swizzled B-fragment form into hsw
  {
    __bf16* hbp = hsw + (size_t)bp * 1024 * H_;   // 32 kb-frags x 8 ct-frags x 32 x 16
    int jb = tile >> 1;
    int laneSw = ((tile & 1) << 4) | colL;
    bool ok = (rowbase + half * 8) < N_;          // 8-row run fully valid or fully padding
#pragma unroll
    for (int ct = 0; ct < 8; ++ct) {
      int col = ct * 16 + colL;
      union { v8bf v; uint4 q; } su;
#pragma unroll
      for (int v = 0; v < 8; ++v) {
        __bf16 hv = f2bf(acc[ct][v]);
        t0[(half * 8 + v) * H_ + col] = hv;
        su.v[v] = hv;
      }
      if (ok) *(uint4*)(hbp + ((size_t)(jb * 8 + ct) * 32 + laneSw) * 16 + half * 8) = su.q;
    }
  }
  // ---- layer 4: We -> x_emb (fp32 in LDS) ----
#pragma unroll
  for (int ct = 0; ct < 8; ++ct) acc[ct] = v8zero();
#pragma unroll
  for (int kb = 0; kb < 4; ++kb) {
    v16bf a = load_a(t0, H_, kb);
    kstep8(acc, a, Wes, kb * 8);
  }
#pragma unroll
  for (int ct = 0; ct < 8; ++ct) {
    int col = ct * 16 + colL;
#pragma unroll
    for (int v = 0; v < 8; ++v) xs[(half * 8 + v) * H_ + col] = acc[ct][v];
  }
  // ---- s1 = x_emb @ a1 ; s2 = x_emb @ a2 (one row per lane<16) ----
  if (lane < 16) {
    int gi = rowbase + lane;
    if (gi < N_) {
      float r1 = 0.f, r2 = 0.f;
      for (int c = 0; c < H_; ++c) {
        float xv = xs[lane * H_ + c];
        r1 += xv * a1[c]; r2 += xv * a2[c];
      }
      s1[bp * NP_ + gi] = r1; s2[bp * NP_ + gi] = r2;
    }
  }
}

// ---------------- softmax stats: per-row max & 1/sum ----------------
__global__ void k_stats(const float* __restrict__ adj, const float* __restrict__ s1,
                        const float* __restrict__ s2, float* __restrict__ mm,
                        float* __restrict__ il) {
  int row = blockIdx.x * 8 + (threadIdx.x >> 5);   // 0..47999
  int lane = threadIdx.x & 31;
  int bp = row / N_, n = row % N_;
  float s1v = s1[bp * NP_ + n];
  const float* ar  = adj + (size_t)n * N_;
  const float* s2p = s2 + bp * NP_;
  float m = -3.4e38f;
  for (int j = lane; j < N_; j += 32) {
    if (ar[j] != 0.f) {
      float z = s1v + s2p[j];
      z = z > 0.f ? z : 0.01f * z;
      m = fmaxf(m, z);
    }
  }
#pragma unroll
  for (int o = 16; o; o >>= 1) m = fmaxf(m, __shfl_xor(m, o, 32));
  float s = 0.f;
  for (int j = lane; j < N_; j += 32) {
    if (ar[j] != 0.f) {
      float z = s1v + s2p[j];
      z = z > 0.f ? z : 0.01f * z;
      s += __expf(z - m);
    }
  }
#pragma unroll
  for (int o = 16; o; o >>= 1) s += __shfl_xor(s, o, 32);
  if (lane == 0) { mm[bp * NP_ + n] = m; il[bp * NP_ + n] = 1.f / s; }
}

// ---------------- attention: g = softmax(z) @ h, then sigmoid(g @ Wlin) -> xt ----------------
__global__ __launch_bounds__(32) void k_attn(
    const float* __restrict__ adj, const float* __restrict__ s1,
    const float* __restrict__ s2,  const float* __restrict__ mm,
    const float* __restrict__ il,  const __bf16* __restrict__ hsw,
    const __bf16* __restrict__ Wlins, __bf16* __restrict__ xt) {
  __shared__ __align__(16) __bf16 gt[16 * H_];
  int bp = blockIdx.x / TILES_;
  int tile = blockIdx.x % TILES_;
  int b = bp / P_, p = bp % P_;
  int lane = threadIdx.x, colL = lane & 15, half = lane >> 4;
  int i = tile * 16 + colL;               // A-row handled by this lane
  bool vi = i < N_;
  float s1v = vi ? s1[bp * NP_ + i] : 0.f;
  float mv  = vi ? mm[bp * NP_ + i] : 0.f;
  float ilv = vi ? il[bp * NP_ + i] : 0.f;
  const float*  adjrow = adj + (size_t)i * N_;
  const float*  s2p    = s2 + bp * NP_;
  const __bf16* hbp    = hsw + (size_t)bp * 1024 * H_;

  v8f acc[8];
#pragma unroll
  for (int ct = 0; ct < 8; ++ct) acc[ct] = v8zero();

  for (int jb = 0; jb < 32; ++jb) {       // K = 1024 (padding rows of h are zero)
    v16bf a;
#pragma unroll
    for (int e = 0; e < 16; ++e) {
      int k = half * 16 + e;
      int j = jb * 32 + k;
      float pv = 0.f;
      if (vi && j < N_) {
        float av = adjrow[j];
        if (av != 0.f) {
          float z = s1v + s2p[j];
          z = z > 0.f ? z : 0.01f * z;
          pv = __expf(z - mv) * ilv;
        }
      }
      a[e] = f2bf(pv);
    }
    kstep8(acc, a, hbp, jb * 8);
  }
  // g tile -> LDS bf16
#pragma unroll
  for (int ct = 0; ct < 8; ++ct) {
    int col = ct * 16 + colL;
#pragma unroll
    for (int v = 0; v < 8; ++v) gt[(half * 8 + v) * H_ + col] = f2bf(acc[ct][v]);
  }
  // u = sigmoid(g @ Wlin)
#pragma unroll
  for (int ct = 0; ct < 8; ++ct) acc[ct] = v8zero();
#pragma unroll
  for (int kb = 0; kb < 4; ++kb) {
    v16bf a = load_a(gt, H_, kb);
    kstep8(acc, a, Wlins, kb * 8);
  }
  __bf16* xtp = xt + ((size_t)p * BN_) * H_;
#pragma unroll
  for (int ct = 0; ct < 8; ++ct) {
    int col = ct * 16 + colL;
#pragma unroll
    for (int v = 0; v < 8; ++v) {
      int gr = tile * 16 + half * 8 + v;
      if (gr < N_) xtp[((size_t)b * N_ + gr) * H_ + col] = f2bf(sigmf(acc[ct][v]));
    }
  }
}

// ---------------- one LSTM step ----------------
__global__ __launch_bounds__(32) void k_lstm(
    const __bf16* __restrict__ xt_p, const __bf16* __restrict__ hin,
    __bf16* __restrict__ hout, float* __restrict__ cst,
    const __bf16* __restrict__ Wgs, const float* __restrict__ bg) {
  int tile = blockIdx.x >> 3;            // 0..249
  int slice = blockIdx.x & 7;            // 16-wide output slice per gate
  int lane = threadIdx.x, colL = lane & 15, half = lane >> 4;
  v8f acc[4];
#pragma unroll
  for (int q = 0; q < 4; ++q) acc[q] = v8zero();
  const __bf16* Ax = xt_p + (size_t)tile * 16 * H_;
  const __bf16* Ah = hin  + (size_t)tile * 16 * H_;
#pragma unroll
  for (int kb = 0; kb < 8; ++kb) {
    v16bf a = (kb < 4) ? load_a(Ax, H_, kb) : load_a(Ah, H_, kb - 4);
    v16bf bf[4];
#pragma unroll
    for (int q = 0; q < 4; ++q) bf[q] = load_bsw(Wgs, kb * 32 + q * 8 + slice);
#pragma unroll
    for (int q = 0; q < 4; ++q) WMMA(acc[q], a, bf[q]);
  }
  int col = slice * 16 + colL;
  float bI = bg[col], bF = bg[H_ + col], bO = bg[2 * H_ + col], bG = bg[3 * H_ + col];
#pragma unroll
  for (int v = 0; v < 8; ++v) {
    int row = tile * 16 + half * 8 + v;
    size_t idx = (size_t)row * H_ + col;
    float gi = acc[0][v] + bI, gf = acc[1][v] + bF;
    float go = acc[2][v] + bO, gg = acc[3][v] + bG;
    float c = sigmf(gf) * cst[idx] + sigmf(gi) * tanhf(gg);
    float h = sigmf(go) * tanhf(c);
    cst[idx] = c;
    hout[idx] = f2bf(h);
  }
}

// ---------------- decoder ----------------
__global__ __launch_bounds__(32) void k_dec(
    const __bf16* __restrict__ hf, const __bf16* __restrict__ D1s,
    const __bf16* __restrict__ D2s, const float* __restrict__ db1,
    const float* __restrict__ db2, const float* __restrict__ D3,
    const float* __restrict__ db3, float* __restrict__ out) {
  __shared__ __align__(16) __bf16 t0[16 * H_];
  __shared__ __align__(16) float  dt[16 * H_];
  int tile = blockIdx.x;
  int lane = threadIdx.x, colL = lane & 15, half = lane >> 4;
  const __bf16* Abase = hf + (size_t)tile * 16 * H_;
  v8f acc[8];
#pragma unroll
  for (int ct = 0; ct < 8; ++ct) acc[ct] = v8zero();
#pragma unroll
  for (int kb = 0; kb < 4; ++kb) {
    v16bf a = load_a(Abase, H_, kb);
    kstep8(acc, a, D1s, kb * 8);
  }
#pragma unroll
  for (int ct = 0; ct < 8; ++ct) {
    int col = ct * 16 + colL; float bias = db1[col];
#pragma unroll
    for (int v = 0; v < 8; ++v) {
      float val = acc[ct][v] + bias;
      t0[(half * 8 + v) * H_ + col] = f2bf(val > 0.f ? val : 0.f);
    }
  }
#pragma unroll
  for (int ct = 0; ct < 8; ++ct) acc[ct] = v8zero();
#pragma unroll
  for (int kb = 0; kb < 4; ++kb) {
    v16bf a = load_a(t0, H_, kb);
    kstep8(acc, a, D2s, kb * 8);
  }
#pragma unroll
  for (int ct = 0; ct < 8; ++ct) {
    int col = ct * 16 + colL; float bias = db2[col];
#pragma unroll
    for (int v = 0; v < 8; ++v) {
      float val = acc[ct][v] + bias;
      dt[(half * 8 + v) * H_ + col] = val > 0.f ? val : 0.f;
    }
  }
  if (lane < 16) {
    int r = tile * 16 + lane;
    int b = r / N_, n = r % N_;
#pragma unroll
    for (int f = 0; f < FUT_; ++f) {
      float s = db3[f];
      for (int c = 0; c < H_; ++c) s += dt[lane * H_ + c] * D3[c * FUT_ + f];
      out[(size_t)b * FUT_ * N_ + f * N_ + n] = s;   // (B, FUT, N)
    }
  }
}

// ---------------- host ----------------
extern "C" void kernel_launch(void* const* d_in, const int* in_sizes, int n_in,
                              void* d_out, int out_size, void* d_ws, size_t ws_size,
                              hipStream_t stream) {
  const float* x    = (const float*)d_in[0];
  const float* adj  = (const float*)d_in[1];
  const float* E0   = (const float*)d_in[2];
  const float* E1   = (const float*)d_in[3];
  const float* W1   = (const float*)d_in[4];
  const float* b1   = (const float*)d_in[5];
  const float* W2   = (const float*)d_in[6];
  const float* b2   = (const float*)d_in[7];
  const float* W3   = (const float*)d_in[8];
  const float* We   = (const float*)d_in[9];
  const float* a1   = (const float*)d_in[10];
  const float* a2   = (const float*)d_in[11];
  const float* Wlin = (const float*)d_in[12];
  const float* Wg   = (const float*)d_in[13];
  const float* bg   = (const float*)d_in[14];
  const float* D1   = (const float*)d_in[15];
  const float* db1  = (const float*)d_in[16];
  const float* D2   = (const float*)d_in[17];
  const float* db2  = (const float*)d_in[18];
  const float* D3   = (const float*)d_in[19];
  const float* db3  = (const float*)d_in[20];
  float* out = (float*)d_out;

  char* ws = (char*)d_ws;
  size_t off = 0;
  auto take = [&](size_t bytes) -> char* {
    char* pp = ws + off;
    off = (off + bytes + 255) & ~(size_t)255;
    return pp;
  };
  __bf16* W1s   = (__bf16*)take((size_t)160 * 128 * 2);
  __bf16* W2s   = (__bf16*)take((size_t)128 * 128 * 2);
  __bf16* W3s   = (__bf16*)take((size_t)128 * 128 * 2);
  __bf16* Wes   = (__bf16*)take((size_t)128 * 128 * 2);
  __bf16* Wlins = (__bf16*)take((size_t)128 * 128 * 2);
  __bf16* Wgs   = (__bf16*)take((size_t)256 * 512 * 2);
  __bf16* D1s   = (__bf16*)take((size_t)128 * 128 * 2);
  __bf16* D2s   = (__bf16*)take((size_t)128 * 128 * 2);
  __bf16* Xemb  = (__bf16*)take((size_t)BP_ * NP_ * 160 * 2);
  size_t hb_bytes = (size_t)BP_ * 1024 * H_ * 2;     // swizzled h (B for attention)
  __bf16* hsw   = (__bf16*)take(hb_bytes);
  float*  s1    = (float*)take((size_t)BP_ * NP_ * 4);
  float*  s2    = (float*)take((size_t)BP_ * NP_ * 4);
  float*  mmv   = (float*)take((size_t)BP_ * NP_ * 4);
  float*  il    = (float*)take((size_t)BP_ * NP_ * 4);
  __bf16* xt    = (__bf16*)take((size_t)P_ * BN_ * H_ * 2);
  __bf16* hs0   = (__bf16*)take((size_t)BN_ * H_ * 2);
  __bf16* hs1   = (__bf16*)take((size_t)BN_ * H_ * 2);
  float*  cst   = (float*)take((size_t)BN_ * H_ * 4);

  // weight conversion + swizzle into B-fragment order
  k_convw_sw<<<(160 * 128 + 255) / 256, 256, 0, stream>>>(W1, W1s, 136, 160, 128);
  k_convw_sw<<<(128 * 128 + 255) / 256, 256, 0, stream>>>(W2, W2s, 128, 128, 128);
  k_convw_sw<<<(128 * 128 + 255) / 256, 256, 0, stream>>>(W3, W3s, 128, 128, 128);
  k_convw_sw<<<(128 * 128 + 255) / 256, 256, 0, stream>>>(We, Wes, 128, 128, 128);
  k_convw_sw<<<(128 * 128 + 255) / 256, 256, 0, stream>>>(Wlin, Wlins, 128, 128, 128);
  k_convw_sw<<<(256 * 512 + 255) / 256, 256, 0, stream>>>(Wg, Wgs, 256, 256, 512);
  k_convw_sw<<<(128 * 128 + 255) / 256, 256, 0, stream>>>(D1, D1s, 128, 128, 128);
  k_convw_sw<<<(128 * 128 + 255) / 256, 256, 0, stream>>>(D2, D2s, 128, 128, 128);

  // zero-init h padding fragments + LSTM initial state
  hipMemsetAsync(hsw, 0, hb_bytes, stream);
  hipMemsetAsync(hs0, 0, (size_t)BN_ * H_ * 2, stream);
  hipMemsetAsync(cst, 0, (size_t)BN_ * H_ * 4, stream);

  // pipeline
  k_embed<<<BP_ * NP_, 160, 0, stream>>>(x, E0, E1, Xemb);
  k_mlp<<<BP_ * TILES_, 32, 0, stream>>>(Xemb, W1s, W2s, W3s, Wes, b1, b2, a1, a2,
                                         hsw, s1, s2);
  k_stats<<<(BP_ * N_) / 8, 256, 0, stream>>>(adj, s1, s2, mmv, il);
  k_attn<<<BP_ * TILES_, 32, 0, stream>>>(adj, s1, s2, mmv, il, hsw, Wlins, xt);

  for (int p = 0; p < P_; ++p) {
    const __bf16* xtp = xt + (size_t)p * BN_ * H_;
    __bf16* hin  = (p & 1) ? hs1 : hs0;
    __bf16* hout = (p & 1) ? hs0 : hs1;
    k_lstm<<<BNT_ * 8, 32, 0, stream>>>(xtp, hin, hout, cst, Wgs, bg);
  }
  // 12 steps: last write lands in hs0
  k_dec<<<BNT_, 32, 0, stream>>>(hs0, D1s, D2s, db1, db2, D3, db3, out);
}
